// EdgeGATConv_38182259261835
// MI455X (gfx1250) — matrix operand order repaired and checked by reading, not compiled, not run on hardware
//
#include <hip/hip_runtime.h>

// ---------------------------------------------------------------------------
// EdgeGAT layer for MI455X (gfx1250): bf16 WMMA GEMMs + async-LDS tile
// staging + fp32 graph ops. N=50000, E=200000, IN=H*D=512, H=8, D=64.
// ---------------------------------------------------------------------------

#define HD    512
#define HEADS 8
#define BN_EPS 1e-5f

typedef __attribute__((ext_vector_type(16))) __bf16 v16bf;
typedef __attribute__((ext_vector_type(8)))  float  v8f;

__device__ __forceinline__ unsigned short f2bf(float f) {
    unsigned u = __float_as_uint(f);
    u += 0x7FFFu + ((u >> 16) & 1u);          // round-to-nearest-even
    return (unsigned short)(u >> 16);
}
__device__ __forceinline__ float bf2f(unsigned short h) {
    return __uint_as_float(((unsigned)h) << 16);
}

__device__ __forceinline__ void atomicMaxF(float* addr, float v) {
    if (v >= 0.0f) atomicMax((int*)addr, __float_as_int(v));
    else           atomicMin((unsigned int*)addr, __float_as_uint(v));
}

// async copy of 16 bytes: memory -> LDS, tracked by ASYNCcnt
__device__ __forceinline__ void async_b128_to_lds(const void* lds_ptr,
                                                  const void* gptr) {
    unsigned lds = (unsigned)(size_t)lds_ptr;        // low 32b = LDS offset
    unsigned long long ga = (unsigned long long)(size_t)gptr;
    asm volatile("global_load_async_to_lds_b128 %0, %1, off"
                 :: "v"(lds), "v"(ga) : "memory");
}
__device__ __forceinline__ void wait_async0() {
    asm volatile("s_wait_asynccnt 0x0" ::: "memory");
}

// ------------------------------- elementwise -------------------------------

__global__ void k_f32_to_bf16(const float* __restrict__ src,
                              unsigned short* __restrict__ dst, long long n) {
    long long i = (long long)blockIdx.x * blockDim.x + threadIdx.x;
    if (i < n) dst[i] = f2bf(src[i]);
}

__global__ void k_fill(float* __restrict__ p, float v, long long n) {
    long long i = (long long)blockIdx.x * blockDim.x + threadIdx.x;
    if (i < n) p[i] = v;
}

// ----------------------------------- GEMM ----------------------------------
// C[M, 512] = A[M, Kdim] @ W[512, Kdim]^T (+bias) (+residual)
// MODE 0: A is a dense bf16 matrix.
// MODE 1: row m of A is concat(nf[src[m]], nf[dst[m]], ef[m])  (Kdim = 1536);
//         per-row source pointers hoisted; region choice is K-loop-uniform.
// Block = 256 threads (8 waves). Block tile 128(M) x 64(N); each wave owns a
// 16-row slice and all 64 columns (4 accumulators). K-step = 32.
// Tiles staged with GLOBAL_LOAD_ASYNC_TO_LDS_B128 into double-buffered LDS;
// the k+1 fetch overlaps the 4 WMMAs of step k (sync = s_wait_asynccnt +
// one workgroup barrier per iteration).
template <int MODE, bool OUT_BF16, bool RESID>
__global__ __launch_bounds__(256) void k_gemm(
        const unsigned short* __restrict__ A,
        const unsigned short* __restrict__ nf,
        const unsigned short* __restrict__ ef,
        const int* __restrict__ srcI,
        const int* __restrict__ dstI,
        const unsigned short* __restrict__ W,
        const float* __restrict__ bias,
        const float* __restrict__ resid,
        void* __restrict__ outv,
        int M, int Kdim) {
    __shared__ __align__(16) unsigned short Asub[2][128 * 32];
    __shared__ __align__(16) unsigned short Wsub[2][64 * 32];

    const int t    = threadIdx.x;
    const int lane = t & 31;
    const int wv   = t >> 5;            // 0..7
    const int n0   = blockIdx.x * 64;
    const int m0   = blockIdx.y * 128;
    const int hi   = (lane >> 4) & 1;   // upper half-wave?

    // ---- per-thread A-tile source pointers, hoisted out of the K loop ----
    // thread covers A-tile chunks idx = t and t+256: row = idx/4, seg = idx%4
    const unsigned short* abaseS[2];
    const unsigned short* abaseD[2];
    const unsigned short* abaseE[2];
    int arow[2], aseg[2];
#pragma unroll
    for (int i = 0; i < 2; ++i) {
        const int idx = t + i * 256;
        arow[i] = idx >> 2;
        aseg[i] = idx & 3;
        int gr = m0 + arow[i];
        if (gr > M - 1) gr = M - 1;     // clamp (stores are guarded)
        if (MODE == 0) {
            abaseS[i] = A + (size_t)gr * Kdim;
        } else {
            abaseS[i] = nf + (size_t)srcI[gr] * HD;
            abaseD[i] = nf + (size_t)dstI[gr] * HD;
            abaseE[i] = ef + (size_t)gr * HD;
        }
    }
    // weight tile: one b128 chunk per thread; LDS row = output column n
    const int wn   = t >> 2;            // 0..63
    const int wseg = t & 3;
    const unsigned short* wbase = W + (size_t)(n0 + wn) * Kdim;

    v8f c[4] = {};

    auto asyncTiles = [&](int kk, int buf) {
#pragma unroll
        for (int i = 0; i < 2; ++i) {
            const unsigned short* b;
            int koff;
            if (MODE == 0) {
                b = abaseS[i];
                koff = kk;
            } else {
                const int region = kk >> 9;     // K-loop-uniform
                b = (region == 0) ? abaseS[i] : (region == 1) ? abaseD[i]
                                                              : abaseE[i];
                koff = kk & (HD - 1);
            }
            async_b128_to_lds(&Asub[buf][arow[i] * 32 + aseg[i] * 8],
                              b + koff + aseg[i] * 8);
        }
        async_b128_to_lds(&Wsub[buf][wn * 32 + wseg * 8],
                          wbase + kk + wseg * 8);
    };

    asyncTiles(0, 0);
    int cur = 0;
    for (int kk = 0; kk < Kdim; kk += 32) {
        wait_async0();          // own wave's async copies done
        __syncthreads();        // everyone's copies done
        if (kk + 32 < Kdim) asyncTiles(kk + 32, cur ^ 1);  // overlap WMMAs

        // A fragment (16x32 bf16): lanes 0-15 K 0-15, lanes 16-31 K 16-31
        union Frag { v16bf v; uint4 u[2]; };
        Frag af, b0, b1, b2, b3;
        const unsigned short* ap =
            &Asub[cur][(wv * 16 + (lane & 15)) * 32 + hi * 16];
        af.u[0] = *(const uint4*)ap;
        af.u[1] = *(const uint4*)(ap + 8);

        // preload all 4 B fragments -> WMMAs can issue back-to-back
        // B fragment (32x16 bf16): lanes 0-15 K 0-15, lanes 16-31 K 16-31
        const unsigned short* bp =
            &Wsub[cur][(lane & 15) * 32 + hi * 16];
        b0.u[0] = *(const uint4*)(bp);              b0.u[1] = *(const uint4*)(bp + 8);
        b1.u[0] = *(const uint4*)(bp + 16 * 32);    b1.u[1] = *(const uint4*)(bp + 16 * 32 + 8);
        b2.u[0] = *(const uint4*)(bp + 32 * 32);    b2.u[1] = *(const uint4*)(bp + 32 * 32 + 8);
        b3.u[0] = *(const uint4*)(bp + 48 * 32);    b3.u[1] = *(const uint4*)(bp + 48 * 32 + 8);

        c[0] = __builtin_amdgcn_wmma_f32_16x16x32_bf16(
            false, af.v, false, b0.v, (short)0, c[0], false, false);
        c[1] = __builtin_amdgcn_wmma_f32_16x16x32_bf16(
            false, af.v, false, b1.v, (short)0, c[1], false, false);
        c[2] = __builtin_amdgcn_wmma_f32_16x16x32_bf16(
            false, af.v, false, b2.v, (short)0, c[2], false, false);
        c[3] = __builtin_amdgcn_wmma_f32_16x16x32_bf16(
            false, af.v, false, b3.v, (short)0, c[3], false, false);

        cur ^= 1;   // alternating buffers: one barrier per iteration suffices
    }

    // ---- store: C layout lanes 0-15 -> M=i, lanes 16-31 -> M=i+8 ----
    const int mrow_base = m0 + wv * 16 + hi * 8;
    const int ncol_base = n0 + (lane & 15);
#pragma unroll
    for (int nt = 0; nt < 4; ++nt) {
        const int gn = ncol_base + nt * 16;
        const float b = bias[gn];
#pragma unroll
        for (int i = 0; i < 8; ++i) {
            const int gm = mrow_base + i;
            if (gm < M) {
                float val = c[nt][i] + b;
                if (RESID) val += resid[(size_t)gm * HD + gn];
                if (OUT_BF16)
                    ((unsigned short*)outv)[(size_t)gm * HD + gn] = f2bf(val);
                else
                    ((float*)outv)[(size_t)gm * HD + gn] = val;
            }
        }
    }
}

// ------------------------- attention logits (per edge) ---------------------
// One wave per edge. Lane covers 16 contiguous channels (head = lane/4);
// shuffle-reduce groups of 4 lanes -> 8 head logits; atomic segment-max.
__global__ void k_logits(const unsigned short* __restrict__ qb,
                         const unsigned short* __restrict__ kb,
                         const int* __restrict__ dstI,
                         float* __restrict__ abuf,
                         float* __restrict__ amax,
                         int E, float scaler) {
    const int lane = threadIdx.x & 31;
    const int e    = blockIdx.x * 8 + (threadIdx.x >> 5);
    if (e >= E) return;
    const int d = dstI[e];

    union { uint4 u[2]; unsigned short s[16]; } qa, ka;
    const unsigned short* qp = qb + (size_t)d * HD + lane * 16;
    const unsigned short* kp = kb + (size_t)e * HD + lane * 16;
    qa.u[0] = *(const uint4*)qp;       qa.u[1] = *(const uint4*)(qp + 8);
    ka.u[0] = *(const uint4*)kp;       ka.u[1] = *(const uint4*)(kp + 8);

    float acc = 0.0f;
#pragma unroll
    for (int i = 0; i < 16; ++i) acc += bf2f(qa.s[i]) * bf2f(ka.s[i]);
    acc += __shfl_xor(acc, 1, 32);
    acc += __shfl_xor(acc, 2, 32);

    if ((lane & 3) == 0) {
        const int h = lane >> 2;
        const float a = acc * scaler;
        abuf[(size_t)e * HEADS + h] = a;
        atomicMaxF(&amax[(size_t)d * HEADS + h], a);
    }
}

__global__ void k_expsum(float* __restrict__ abuf,
                         const float* __restrict__ amax,
                         float* __restrict__ asum,
                         const int* __restrict__ dstI, int E) {
    const int i = blockIdx.x * blockDim.x + threadIdx.x;
    if (i >= E * HEADS) return;
    const int e = i >> 3, h = i & 7;
    const int d = dstI[e];
    const float ae = __expf(abuf[i] - amax[d * HEADS + h]);
    abuf[i] = ae;
    atomicAdd(&asum[d * HEADS + h], ae);
}

// weighted scatter-sum of v into dst rows (wave per edge)
__global__ void k_scatter(const unsigned short* __restrict__ vb,
                          const float* __restrict__ abuf,
                          const float* __restrict__ asum,
                          const int* __restrict__ dstI,
                          float* __restrict__ agg, int E) {
    const int lane = threadIdx.x & 31;
    const int e    = blockIdx.x * 8 + (threadIdx.x >> 5);
    if (e >= E) return;
    const int d = dstI[e];
    const int h = lane >> 2;
    const float w = abuf[(size_t)e * HEADS + h] / asum[(size_t)d * HEADS + h];

    union { uint4 u[2]; unsigned short s[16]; } va;
    const unsigned short* vp = vb + (size_t)e * HD + lane * 16;
    va.u[0] = *(const uint4*)vp;  va.u[1] = *(const uint4*)(vp + 8);

    float* op = agg + (size_t)d * HD + lane * 16;
#pragma unroll
    for (int i = 0; i < 16; ++i) atomicAdd(op + i, bf2f(va.s[i]) * w);
}

// ------------------------------- batch norm --------------------------------

__global__ void k_bnstats(const float* __restrict__ pre,
                          float* __restrict__ cs, float* __restrict__ cq,
                          int n, int rowsPerBlock) {
    const int c  = threadIdx.x;                 // columns c and c+256
    const int r0 = blockIdx.x * rowsPerBlock;
    const int r1 = min(n, r0 + rowsPerBlock);
    float s0 = 0, q0 = 0, s1 = 0, q1 = 0;
    for (int r = r0; r < r1; ++r) {
        const float* row = pre + (size_t)r * HD;
        const float x = row[c];        s0 += x; q0 += x * x;
        const float y = row[c + 256];  s1 += y; q1 += y * y;
    }
    atomicAdd(&cs[c], s0);        atomicAdd(&cq[c], q0);
    atomicAdd(&cs[c + 256], s1);  atomicAdd(&cq[c + 256], q1);
}

__global__ void k_bnapply(const float* __restrict__ pre,
                          const float* __restrict__ cs,
                          const float* __restrict__ cq,
                          const float* __restrict__ gamma,
                          const float* __restrict__ beta,
                          float* __restrict__ out,
                          long long total, float invN) {
    const long long i = (long long)blockIdx.x * blockDim.x + threadIdx.x;
    if (i >= total) return;
    const int c = (int)(i & (HD - 1));
    const float mean = cs[c] * invN;
    const float var  = cq[c] * invN - mean * mean;
    out[i] = (pre[i] - mean) * rsqrtf(var + BN_EPS) * gamma[c] + beta[c];
}

// --------------------------------- launch ----------------------------------

extern "C" void kernel_launch(void* const* d_in, const int* in_sizes, int n_in,
                              void* d_out, int out_size, void* d_ws, size_t ws_size,
                              hipStream_t stream) {
    const float* node_feat = (const float*)d_in[0];
    const float* edge_feat = (const float*)d_in[1];
    const int*   srcI      = (const int*)d_in[2];
    const int*   dstI      = (const int*)d_in[3];
    const float* Wq = (const float*)d_in[4];   const float* bq = (const float*)d_in[5];
    const float* Wr = (const float*)d_in[6];   const float* br = (const float*)d_in[7];
    const float* Wk = (const float*)d_in[8];   const float* bk = (const float*)d_in[9];
    const float* Wv = (const float*)d_in[10];  const float* bv = (const float*)d_in[11];
    const float* Wo = (const float*)d_in[12];  const float* bo = (const float*)d_in[13];
    const float* gamma = (const float*)d_in[14];
    const float* beta  = (const float*)d_in[15];

    const int N = in_sizes[0] / HD;
    const int E = in_sizes[2];

    // ---- workspace carve (256-B aligned offsets) ----
    char* ws = (char*)d_ws;
    size_t off = 0;
    auto take = [&](size_t bytes) -> char* {
        char* p = ws + off;
        off += (bytes + 255) & ~(size_t)255;
        return p;
    };
    unsigned short* nf16  = (unsigned short*)take((size_t)N * HD * 2);
    unsigned short* ef16  = (unsigned short*)take((size_t)E * HD * 2);
    unsigned short* Wr16  = (unsigned short*)take((size_t)HD * 3 * HD * 2);
    unsigned short* Wq16  = (unsigned short*)take((size_t)HD * HD * 2);
    unsigned short* Wk16  = (unsigned short*)take((size_t)HD * HD * 2);
    unsigned short* Wv16  = (unsigned short*)take((size_t)HD * HD * 2);
    unsigned short* Wo16  = (unsigned short*)take((size_t)HD * HD * 2);
    unsigned short* rbuf  = (unsigned short*)take((size_t)E * HD * 2);
    unsigned short* qbuf  = (unsigned short*)take((size_t)N * HD * 2);
    unsigned short* kbuf  = (unsigned short*)take((size_t)E * HD * 2);
    unsigned short* vbuf  = (unsigned short*)take((size_t)E * HD * 2);
    float* abuf   = (float*)take((size_t)E * HEADS * 4);
    float* amax   = (float*)take((size_t)N * HEADS * 4);
    float* asum   = (float*)take((size_t)N * HEADS * 4);
    float* agg    = (float*)take((size_t)N * HD * 4);
    float* colsum = (float*)take(HD * 4);
    float* colsq  = (float*)take(HD * 4);
    // buffers dead after scatter are reused:
    unsigned short* agg16 = kbuf;          // N*HD*2 <= E*HD*2
    float*          preBN = (float*)vbuf;  // N*HD*4 <= E*HD*2

    const int THR = 256;
    auto blks = [&](long long n) { return (unsigned)((n + THR - 1) / THR); };

    // 1) f32 -> bf16 conversions
    k_f32_to_bf16<<<blks((long long)N * HD), THR, 0, stream>>>(node_feat, nf16, (long long)N * HD);
    k_f32_to_bf16<<<blks((long long)E * HD), THR, 0, stream>>>(edge_feat, ef16, (long long)E * HD);
    k_f32_to_bf16<<<blks((long long)HD * 3 * HD), THR, 0, stream>>>(Wr, Wr16, (long long)HD * 3 * HD);
    k_f32_to_bf16<<<blks((long long)HD * HD), THR, 0, stream>>>(Wq, Wq16, (long long)HD * HD);
    k_f32_to_bf16<<<blks((long long)HD * HD), THR, 0, stream>>>(Wk, Wk16, (long long)HD * HD);
    k_f32_to_bf16<<<blks((long long)HD * HD), THR, 0, stream>>>(Wv, Wv16, (long long)HD * HD);
    k_f32_to_bf16<<<blks((long long)HD * HD), THR, 0, stream>>>(Wo, Wo16, (long long)HD * HD);

    // 2) init accumulators
    k_fill<<<blks((long long)N * HEADS), THR, 0, stream>>>(amax, -__builtin_inff(), (long long)N * HEADS);
    k_fill<<<blks((long long)N * HEADS), THR, 0, stream>>>(asum, 0.0f, (long long)N * HEADS);
    k_fill<<<blks((long long)N * HD), THR, 0, stream>>>(agg, 0.0f, (long long)N * HD);
    k_fill<<<blks(HD), THR, 0, stream>>>(colsum, 0.0f, HD);
    k_fill<<<blks(HD), THR, 0, stream>>>(colsq, 0.0f, HD);

    // 3) GEMMs (bf16 WMMA, f32 accumulate)
    const dim3 gblk(256);
    // r = cat(nf[src], nf[dst], ef) @ Wr^T + br      [E, 512], K=1536 (fused gather)
    k_gemm<1, true, false><<<dim3(8, (E + 127) / 128), gblk, 0, stream>>>(
        nullptr, nf16, ef16, srcI, dstI, Wr16, br, nullptr, rbuf, E, 3 * HD);
    // q = nf @ Wq^T + bq                              [N, 512]
    k_gemm<0, true, false><<<dim3(8, (N + 127) / 128), gblk, 0, stream>>>(
        nf16, nullptr, nullptr, nullptr, nullptr, Wq16, bq, nullptr, qbuf, N, HD);
    // k = r @ Wk^T + bk ; v = r @ Wv^T + bv           [E, 512]
    k_gemm<0, true, false><<<dim3(8, (E + 127) / 128), gblk, 0, stream>>>(
        rbuf, nullptr, nullptr, nullptr, nullptr, Wk16, bk, nullptr, kbuf, E, HD);
    k_gemm<0, true, false><<<dim3(8, (E + 127) / 128), gblk, 0, stream>>>(
        rbuf, nullptr, nullptr, nullptr, nullptr, Wv16, bv, nullptr, vbuf, E, HD);

    // 4) attention: logits -> segment max -> exp/segment sum -> scatter
    k_logits<<<(E + 7) / 8, THR, 0, stream>>>(qbuf, kbuf, dstI, abuf, amax, E, 0.125f);
    k_expsum<<<blks((long long)E * HEADS), THR, 0, stream>>>(abuf, amax, asum, dstI, E);
    k_scatter<<<(E + 7) / 8, THR, 0, stream>>>(vbuf, abuf, asum, dstI, agg, E);

    // 5) output projection + residual (f32 out), then batch norm
    k_f32_to_bf16<<<blks((long long)N * HD), THR, 0, stream>>>(agg, agg16, (long long)N * HD);
    k_gemm<0, false, true><<<dim3(8, (N + 127) / 128), gblk, 0, stream>>>(
        agg16, nullptr, nullptr, nullptr, nullptr, Wo16, bo, node_feat, preBN, N, HD);

    k_bnstats<<<256, 256, 0, stream>>>(preBN, colsum, colsq, N, (N + 255) / 256);
    k_bnapply<<<blks((long long)N * HD), THR, 0, stream>>>(
        preBN, colsum, colsq, gamma, beta, (float*)d_out, (long long)N * HD, 1.0f / (float)N);

    (void)n_in; (void)out_size; (void)ws_size;
}
